// GGCN_relation_48292612276722
// MI455X (gfx1250) — compile-verified, matrix operand order checked by reading, not run on hardware
//
#include <hip/hip_runtime.h>
#include <hip/hip_bf16.h>

typedef __attribute__((ext_vector_type(16))) __bf16 v16bf;
typedef __attribute__((ext_vector_type(2)))  __bf16 v2bf;
typedef __attribute__((ext_vector_type(8)))  float  v8f;
typedef __attribute__((ext_vector_type(4)))  int    v4i;

#define APAD 40   // bf16 elems per LDS row: 80B, 16B-aligned, bank-spread

// Native f32->bf16 (RNE) — backend emits v_cvt_pk_bf16_f32.
__device__ __forceinline__ unsigned int pack2(float a, float b) {
  union { v2bf v; unsigned int u; } u;
  u.v[0] = (__bf16)a; u.v[1] = (__bf16)b;
  return u.u;
}

union FragBF { v16bf v; v4i i2[2]; };

// Generic tiled GEMM:  C_b = act(Ahat_b @ B_b + bias), bf16 WMMA / f32 accum.
// Block = 256 threads = 8 waves; 128x128 C tile; K step 32.
// DIAG: Ahat = A + I (GCN self-loops).  GUARD: K not multiple of 32 (K even).
// Register double-buffer (global prefetch) + LDS ping-pong (1 barrier/step).
template<bool DIAG, bool GUARD>
__global__ __launch_bounds__(256)
void gemm_bf16(const float* __restrict__ A, long long sAb, int lda,
               const float* __restrict__ B, long long sBb, int ldb,
               float* __restrict__ C, long long sCb, int ldc, int colOff,
               const float* __restrict__ bias, int doRelu, int mapMode,
               int N, int K)
{
  __shared__ alignas(16) unsigned short As[2][128 * APAD];
  __shared__ alignas(16) unsigned short Bs[2][128 * APAD];

  const float* Ab = A + (long long)blockIdx.z * sAb;
  const float* Bb = B + (long long)blockIdx.z * sBb;
  float*       Cb = C + (long long)blockIdx.z * sCb;
  (void)N;

  const int rowBase = blockIdx.y * 128;
  const int colBase = blockIdx.x * 128;

  const int tid   = threadIdx.x;
  const int wave  = tid >> 5;
  const int lane  = tid & 31;
  const int waveM = wave & 3;            // 4 row groups of 32
  const int waveN = wave >> 2;           // 2 col groups of 64
  const int lm    = lane & 15;
  const int khalf = (lane >> 4) << 3;    // 0 or 8

  // A staging: row-major float4 loads (coalesced), packed b64 LDS stores.
  const int arow0 = tid >> 3;            // + it*32
  const int akk   = (tid & 7) * 4;
  // B staging: two adjacent k-rows per thread -> packed b32 transposed stores.
  const int bkp   = tid >> 5;            // row-pair index, + it*8  (k = 2*pair)
  const int bnq   = (tid & 31) * 4;

  v8f acc[2][4];
#pragma unroll
  for (int i = 0; i < 2; i++)
#pragma unroll
    for (int j = 0; j < 4; j++)
      acc[i][j] = (v8f){0.f,0.f,0.f,0.f,0.f,0.f,0.f,0.f};

  float4 aR[4], bR0[2], bR1[2];
  const float4 z4 = make_float4(0.f, 0.f, 0.f, 0.f);

  auto loadTiles = [&](int ks) {
    const int k0 = ks << 5;
#pragma unroll
    for (int it = 0; it < 4; it++) {
      int gk = k0 + akk;
      aR[it] = (!GUARD || gk < K)
          ? *(const float4*)(&Ab[(long long)(rowBase + arow0 + it * 32) * lda + gk])
          : z4;
    }
#pragma unroll
    for (int it = 0; it < 2; it++) {
      int gk = k0 + (bkp + it * 8) * 2;
      bR0[it] = (!GUARD || gk < K)
          ? *(const float4*)(&Bb[(long long)gk * ldb + (colBase + bnq)]) : z4;
      bR1[it] = (!GUARD || gk + 1 < K)
          ? *(const float4*)(&Bb[(long long)(gk + 1) * ldb + (colBase + bnq)]) : z4;
    }
  };

  auto storeTiles = [&](int buf, int ks) {
    const int k0 = ks << 5;
#pragma unroll
    for (int it = 0; it < 4; it++) {
      int r = arow0 + it * 32;
      float f[4] = {aR[it].x, aR[it].y, aR[it].z, aR[it].w};
      if (DIAG) {
        int d = (rowBase + r) - (k0 + akk);
        if ((unsigned)d < 4u) f[d] += 1.f;
      }
      uint2 p = make_uint2(pack2(f[0], f[1]), pack2(f[2], f[3]));
      *(uint2*)(&As[buf][r * APAD + akk]) = p;          // ds_store_b64
    }
#pragma unroll
    for (int it = 0; it < 2; it++) {
      int k = (bkp + it * 8) * 2;
      float l0[4] = {bR0[it].x, bR0[it].y, bR0[it].z, bR0[it].w};
      float l1[4] = {bR1[it].x, bR1[it].y, bR1[it].z, bR1[it].w};
#pragma unroll
      for (int e = 0; e < 4; e++) {     // (k,k+1) pair contiguous in Bs[n][*]
        *(unsigned int*)(&Bs[buf][(bnq + e) * APAD + k]) = pack2(l0[e], l1[e]);
      }
    }
  };

  const int nk = (K + 31) >> 5;
  loadTiles(0);
  storeTiles(0, 0);
  int cur = 0;
  for (int ks = 0; ks < nk; ks++) {
    __syncthreads();                    // tile `cur` visible; prev reads retired
    const bool more = (ks + 1 < nk);
    if (more) loadTiles(ks + 1);        // global prefetch overlaps WMMAs

    FragBF a[2], b[4];
#pragma unroll
    for (int i = 0; i < 2; i++) {
      const unsigned short* p = &As[cur][(waveM * 32 + i * 16 + lm) * APAD];
      a[i].i2[0] = *(const v4i*)(p + khalf);        // K = khalf..khalf+7
      a[i].i2[1] = *(const v4i*)(p + 16 + khalf);   // K = 16+khalf..
    }
#pragma unroll
    for (int j = 0; j < 4; j++) {
      const unsigned short* p = &Bs[cur][(waveN * 64 + j * 16 + lm) * APAD];
      b[j].i2[0] = *(const v4i*)(p + khalf);
      b[j].i2[1] = *(const v4i*)(p + 16 + khalf);
    }
#pragma unroll
    for (int i = 0; i < 2; i++)
#pragma unroll
      for (int j = 0; j < 4; j++)
        acc[i][j] = __builtin_amdgcn_wmma_f32_16x16x32_bf16(
            false, a[i].v, false, b[j].v, (short)0, acc[i][j], false, false);

    if (more) storeTiles(cur ^ 1, ks + 1);   // convert+store next tile
    cur ^= 1;
  }

  // Epilogue: bias, ReLU, optional row remap, column offset into wide buffer.
#pragma unroll
  for (int i = 0; i < 2; i++) {
#pragma unroll
    for (int j = 0; j < 4; j++) {
      int colN = colBase + waveN * 64 + j * 16 + lm;
      float bv = bias ? bias[colN] : 0.f;
#pragma unroll
      for (int r = 0; r < 8; r++) {
        int row = rowBase + waveM * 32 + i * 16 + r + ((lane >> 4) << 3);
        float v = acc[i][j][r] + bv;
        if (doRelu) v = fmaxf(v, 0.f);
        int orow = row;
        if (mapMode == 1) {           // stacked [a;o] means -> cat rows
          int s = row, off = 0;
          if (s >= 2048) { s -= 2048; off = 8; }
          orow = (s >> 3) * 16 + off + (s & 7);
        }
        Cb[(long long)orow * ldc + colOff + colN] = v;
      }
    }
  }
}

// Per-segment MHA (S=8, H=4, dh=128) with the mean folded before Wp:
// out[g*2048+s][c] = (1/8) sum_si sum_sj softmax(qk/sqrt(dh))[si][sj] * v[sj][c]
__global__ __launch_bounds__(256)
void attn_mean(const float* __restrict__ Q, const float* __restrict__ Km,
               const float* __restrict__ V, float* __restrict__ Out)
{
  __shared__ float qs[8][512], ks[8][512], vs[8][512];
  __shared__ float sc[4][8][8];
  __shared__ float colsum[4][8];
  const int s = blockIdx.x;   // segment 0..2047
  const int g = blockIdx.y;   // 0 = seg_a (i-major), 1 = seg_o (j-major)
  const int tid = threadIdx.x;

  for (int idx = tid; idx < 8 * 512; idx += 256) {
    int j = idx >> 9, c = idx & 511;
    int grow;
    if (g == 0) grow = s * 8 + j;
    else { int bt = s >> 3, jj = s & 7; grow = bt * 64 + j * 8 + jj; }
    long long o = (long long)grow * 512 + c;
    qs[j][c] = Q[o]; ks[j][c] = Km[o]; vs[j][c] = V[o];
  }
  __syncthreads();
  {
    int h = tid >> 6, p = tid & 63, si = p >> 3, sj = p & 7;
    const float* qp = &qs[si][h * 128];
    const float* kp = &ks[sj][h * 128];
    float d = 0.f;
#pragma unroll 4
    for (int c = 0; c < 128; c++) d += qp[c] * kp[c];
    sc[h][si][sj] = d * 0.08838834764831845f;   // 1/sqrt(128)
  }
  __syncthreads();
  if (tid < 32) {
    int h = tid >> 3, si = tid & 7;
    float m = sc[h][si][0];
#pragma unroll
    for (int j = 1; j < 8; j++) m = fmaxf(m, sc[h][si][j]);
    float sum = 0.f;
#pragma unroll
    for (int j = 0; j < 8; j++) { float e = __expf(sc[h][si][j] - m); sc[h][si][j] = e; sum += e; }
    float inv = 1.f / sum;
#pragma unroll
    for (int j = 0; j < 8; j++) sc[h][si][j] *= inv;
  }
  __syncthreads();
  if (tid < 32) {
    int h = tid >> 3, sj = tid & 7;
    float t = 0.f;
#pragma unroll
    for (int si = 0; si < 8; si++) t += sc[h][si][sj];
    colsum[h][sj] = t * 0.125f;                 // fold the mean over S
  }
  __syncthreads();
  for (int c = tid; c < 512; c += 256) {
    int h = c >> 7;
    float v = 0.f;
#pragma unroll
    for (int sj = 0; sj < 8; sj++) v += colsum[h][sj] * vs[sj][c];
    Out[((long long)(g * 2048 + s)) * 512 + c] = v;
  }
}

extern "C" void kernel_launch(void* const* d_in, const int* in_sizes, int n_in,
                              void* d_out, int out_size, void* d_ws, size_t ws_size,
                              hipStream_t stream) {
  (void)in_sizes; (void)n_in; (void)out_size; (void)ws_size;
  const float* Ao   = (const float*)d_in[0];
  const float* srco = (const float*)d_in[1];
  const float* Ar   = (const float*)d_in[2];
  const float* srcr = (const float*)d_in[3];
  const float* Wo1  = (const float*)d_in[4];
  const float* Wo2  = (const float*)d_in[5];
  const float* Wr1  = (const float*)d_in[6];
  const float* Wr2  = (const float*)d_in[7];
  const float* Wq   = (const float*)d_in[8];
  const float* bq   = (const float*)d_in[9];
  const float* Wk   = (const float*)d_in[10];
  const float* bk   = (const float*)d_in[11];
  const float* Wv   = (const float*)d_in[12];
  const float* bv   = (const float*)d_in[13];
  const float* Wp   = (const float*)d_in[14];
  const float* bp   = (const float*)d_in[15];
  const float* We   = (const float*)d_in[16];
  const float* be   = (const float*)d_in[17];
  float* out = (float*)d_out;
  float* ws  = (float*)d_ws;

  const long long MF = 1024LL * 1024LL;           // floats
  float* cat  = ws;              //  4M floats  (4096 x 1024)
  float* bufA = ws + 4 * MF;     // 16M floats
  float* bufB = ws + 20 * MF;    // 16M floats
  float* bufC = ws + 36 * MF;    //  8M floats  (peak 44M floats = 176MB)

  dim3 blk(256);
  auto grd = [](int M, int N, int Z) { return dim3(N / 128, M / 128, Z); };

  // ---- GCN-O ----
  float* t1 = bufA;   // (4096,1024)
  gemm_bf16<false,false><<<grd(4096,1024,1),blk,0,stream>>>(srco,0,1024, Wo1,0,1024,
      t1,0,1024,0, nullptr,0,0, 1024,1024);
  float* h1 = bufB;   // (8 x 512,1024), relu((Ao+I)@t1)
  gemm_bf16<true,false><<<grd(512,1024,8),blk,0,stream>>>(Ao,512LL*512,512, t1,512LL*1024,1024,
      h1,512LL*1024,1024,0, nullptr,1,0, 1024,512);
  float* t2 = bufA;   // (4096,512)
  gemm_bf16<false,false><<<grd(4096,512,1),blk,0,stream>>>(h1,0,1024, Wo2,0,512,
      t2,0,512,0, nullptr,0,0, 512,1024);
  // out_gcn_o -> right half of cat (rows map identically)
  gemm_bf16<true,false><<<grd(512,512,8),blk,0,stream>>>(Ao,512LL*512,512, t2,512LL*512,512,
      cat,512LL*1024,1024,512, nullptr,1,0, 512,512);

  // ---- GCN-R (dominant: Ar is 8 x 2048 x 2048) ----
  float* u1 = bufA;   // (16384,1024); K=300 zero-padded (K even -> pair guard ok)
  gemm_bf16<false,true><<<grd(16384,1024,1),blk,0,stream>>>(srcr,0,300, Wr1,0,1024,
      u1,0,1024,0, nullptr,0,0, 1024,300);
  float* g1 = bufB;   // relu((Ar+I)@u1)
  gemm_bf16<true,false><<<grd(2048,1024,8),blk,0,stream>>>(Ar,2048LL*2048,2048, u1,2048LL*1024,1024,
      g1,2048LL*1024,1024,0, nullptr,1,0, 1024,2048);
  float* u2 = bufA;   // (16384,512)
  gemm_bf16<false,false><<<grd(16384,512,1),blk,0,stream>>>(g1,0,1024, Wr2,0,512,
      u2,0,512,0, nullptr,0,0, 512,1024);
  float* X = bufA + 8 * MF;  // out_gcn_r flat (16384,512)
  gemm_bf16<true,false><<<grd(2048,512,8),blk,0,stream>>>(Ar,2048LL*2048,2048, u2,2048LL*512,512,
      X,2048LL*512,512,0, nullptr,1,0, 512,2048);

  // ---- MHA: shared Q/K/V for both groupings ----
  float* Qb = bufB;            // (16384,512)
  float* Kb = bufB + 8 * MF;
  float* Vb = bufC;
  gemm_bf16<false,false><<<grd(16384,512,1),blk,0,stream>>>(X,0,512, Wq,0,512,
      Qb,0,512,0, bq,0,0, 512,512);
  gemm_bf16<false,false><<<grd(16384,512,1),blk,0,stream>>>(X,0,512, Wk,0,512,
      Kb,0,512,0, bk,0,0, 512,512);
  gemm_bf16<false,false><<<grd(16384,512,1),blk,0,stream>>>(X,0,512, Wv,0,512,
      Vb,0,512,0, bv,0,0, 512,512);

  float* means = bufA;         // (4096,512): [a-means(2048); o-means(2048)]
  attn_mean<<<dim3(2048,2),blk,0,stream>>>(Qb, Kb, Vb, means);

  // Wp projection with mean already folded; epilogue row-permute into cat[:, :512]
  gemm_bf16<false,false><<<grd(4096,512,1),blk,0,stream>>>(means,0,512, Wp,0,512,
      cat,0,1024,0, bp,0,1, 512,512);

  // ---- Final edge->node projection ----
  gemm_bf16<false,false><<<grd(4096,512,1),blk,0,stream>>>(cat,0,1024, We,0,512,
      out,0,512,0, be,0,0, 512,1024);
}
